// ShiftedWindowAttention_73607149519477
// MI455X (gfx1250) — matrix operand order repaired and checked by reading, not compiled, not run on hardware
//
#include <hip/hip_runtime.h>

typedef __attribute__((ext_vector_type(16))) __bf16 v16bf;
typedef __attribute__((ext_vector_type(8)))  __bf16 v8bf;
typedef __attribute__((ext_vector_type(4)))  __bf16 v4bf;
typedef __attribute__((ext_vector_type(8)))  float  v8f;
typedef __attribute__((ext_vector_type(4)))  float  v4f;
typedef __attribute__((ext_vector_type(4)))  int    v4i;

#define WMMA_BF16(c, a, b)                                                    \
  (c) = __builtin_amdgcn_wmma_f32_16x16x32_bf16(false, (a), false, (b),       \
                                                (short)0, (c), false, false)

static __device__ __forceinline__ v16bf join8(v8bf lo, v8bf hi) {
  return __builtin_shufflevector(lo, hi, 0, 1, 2, 3, 4, 5, 6, 7,
                                         8, 9, 10, 11, 12, 13, 14, 15);
}

// A-matrix fragment (16x32 bf16), documented CDNA5 A interleave:
// lanes 0-15 hold K[0..7] + K[16..23]; lanes 16-31 hold K[8..15] + K[24..31].
static __device__ __forceinline__ v16bf
load_a_frag(const __bf16* base, int row, int stride, int k0, int lane) {
  const int kofs = (lane >= 16) ? 8 : 0;
  const __bf16* p = base + row * stride + k0 + kofs;
  v8bf lo = *(const v8bf*)(p);
  v8bf hi = *(const v8bf*)(p + 16);
  return join8(lo, hi);
}

// B-matrix fragment (32x16 bf16), column n contiguous (row-major B^T):
// lanes 0-15 hold K[0..7] + K[16..23], lanes 16-31 hold K[8..15] + K[24..31] of col ncol.
static __device__ __forceinline__ v16bf
load_bt_frag(const __bf16* base, int ncol, int stride, int k0, int lane) {
  const int kofs = (lane >= 16) ? 8 : 0;
  const __bf16* p = base + ncol * stride + k0 + kofs;
  v8bf lo = *(const v8bf*)(p);
  v8bf hi = *(const v8bf*)(p + 16);
  return join8(lo, hi);
}

// ---------------------------------------------------------------------------
// ws layout: [0)      qkv_w bf16 384*128, q rows pre-scaled by 32^-0.5 (98304B)
//            [98304)  proj_w bf16 128*128                            (32768 B)
//            [131072) bm f32 [4 var][4 h][64 i][16 jc][4 nt2]       (262144 B)
//            [393216) qkv_b f32 [384], q part pre-scaled             (1536 B)
// ---------------------------------------------------------------------------
__global__ void swin_prep(const float* __restrict__ qkv_w,
                          const float* __restrict__ proj_w,
                          const float* __restrict__ qkv_b,
                          const float* __restrict__ reltab,  // [169][4]
                          char* __restrict__ ws) {
  __bf16* qkvw_bf = (__bf16*)ws;
  __bf16* projw_bf = (__bf16*)(ws + 98304);
  float*  bm = (float*)(ws + 131072);
  float*  qkvb_s = (float*)(ws + 393216);
  const float qscale = 0.17677669529663687f;  // 32^-0.5

  const int NW = 384 * 128 + 128 * 128;  // 65536 weight elements
  const int NB = 4 * 4 * 64 * 64;        // 65536 bm elements
  for (int id = blockIdx.x * blockDim.x + threadIdx.x; id < NW + NB + 384;
       id += gridDim.x * blockDim.x) {
    if (id < 384 * 128) {
      float w = qkv_w[id];
      if (id < 128 * 128) w *= qscale;            // fold q scaling into W
      qkvw_bf[id] = (__bf16)w;
    } else if (id < NW) {
      projw_bf[id - 384 * 128] = (__bf16)proj_w[id - 384 * 128];
    } else if (id < NW + NB) {
      const int e = id - NW;
      const int v = e >> 14;         // variant: (wi==15)*2 | (wj==15)
      const int h = (e >> 12) & 3;   // head
      const int i = (e >> 6) & 63;   // query token
      const int j = e & 63;          // key token
      float val;
      if (j >= 49) {
        val = -1.0e30f;              // padded key -> masked out
      } else if (i >= 49) {
        val = 0.f;                   // padded query row: unused
      } else {
        const int ih = i / 7, iw = i % 7, jh = j / 7, jw = j % 7;
        val = reltab[((ih - jh + 6) * 13 + (iw - jw + 6)) * 4 + h];
        const int wi15 = v >> 1, wj15 = v & 1;
        const int rih = wi15 ? (ih < 4 ? 1 : 2) : 0;
        const int riw = wj15 ? (iw < 4 ? 1 : 2) : 0;
        const int rjh = wi15 ? (jh < 4 ? 1 : 2) : 0;
        const int rjw = wj15 ? (jw < 4 ? 1 : 2) : 0;
        if (rih * 3 + riw != rjh * 3 + rjw) val -= 100.f;
      }
      // interleaved layout: [v][h][i][j&15][j>>4] -> v4f loads in stage 3
      bm[(((v * 4 + h) * 64 + i) * 16 + (j & 15)) * 4 + (j >> 4)] = val;
    } else {
      const int e = id - NW - NB;    // bias, q part pre-scaled
      qkvb_s[e] = qkv_b[e] * (e < 128 ? qscale : 1.f);
    }
  }
}

// ---------------------------------------------------------------------------
// Main fused kernel: one workgroup (256 thr = 8 wave32) per (batch, window).
// ---------------------------------------------------------------------------
__global__ __launch_bounds__(256)
void swin_window_attn_fused(const float* __restrict__ x,
                            const float* __restrict__ proj_b,  // [128]
                            const char* __restrict__ ws,
                            float* __restrict__ out) {
  constexpr int H = 112, W = 112, C = 128, SH = 3, NTOK = 49;
  const __bf16* qkvw_bf = (const __bf16*)ws;               // [384][128]
  const __bf16* projw_bf = (const __bf16*)(ws + 98304);    // [128][128]
  const float*  bm = (const float*)(ws + 131072);          // interleaved
  const float*  qkvb_s = (const float*)(ws + 393216);      // [384]

  __shared__ __align__(32) __bf16 smem[32768];             // 64 KB
  __bf16* xa = smem;                       // [64][128] x tile; later attn-out
  __bf16* qk = smem + 64 * 128;            // [64][256] q|k ; later P / rowoff
  __bf16* vt = smem + 64 * 128 + 64 * 256; // [4][32][64] v transposed

  const int tid = threadIdx.x;
  const int lane = tid & 31;
  const int wvu = __builtin_amdgcn_readfirstlane(tid >> 5); // scalar wave id
  const int lrow = lane & 15;
  const int hi16 = (lane >= 16) ? 8 : 0;

  const int blk = blockIdx.x;
  const int b = blk >> 8;                  // batch
  const int wi = (blk & 255) >> 4;         // window row 0..15
  const int wj = blk & 15;                 // window col 0..15

  // ---------- Stage 1: shifted window -> LDS (bf16), rows 49..63 zero ----
  // wave handles rows wv+8k (scalar row -> SALU address, b128 loads).
#pragma unroll
  for (int k = 0; k < 8; ++k) {
    const int m = wvu + 8 * k;             // scalar
    v4f f = {};
    if (m < NTOK) {
      const int gh = (wi * 7 + (m / 7) + SH) % H;
      const int gw = (wj * 7 + (m % 7) + SH) % W;
      f = *(const v4f*)(x + ((((size_t)b * H + gh) * W + gw) * C + lane * 4));
    }
    v4bf o;
#pragma unroll
    for (int j = 0; j < 4; ++j) o[j] = (__bf16)f[j];
    *(v4bf*)(xa + m * 128 + lane * 4) = o;
  }
  __syncthreads();

  // ---------- Stage 2: QKV = x @ W^T + b  (B-fragments reused over M) ----
  for (int t = 0; t < 3; ++t) {
    const int nt = __builtin_amdgcn_readfirstlane(wvu * 3 + t);  // 0..23
    v16bf B[4];
#pragma unroll
    for (int kk = 0; kk < 4; ++kk)
      B[kk] = load_bt_frag(qkvw_bf, nt * 16 + lrow, 128, kk * 32, lane);
    const float bias = qkvb_s[nt * 16 + lrow];

#pragma unroll
    for (int mt = 0; mt < 4; ++mt) {
      v8f acc = {};
#pragma unroll
      for (int kk = 0; kk < 4; ++kk) {
        v16bf a = load_a_frag(xa, mt * 16 + lrow, 128, kk * 32, lane);
        WMMA_BF16(acc, a, B[kk]);
      }
      if (nt < 16) {                       // q or k: row-major scatter
        const int cofs = ((nt >= 8) ? 128 : 0) + (nt & 7) * 16 + lrow;
#pragma unroll
        for (int r = 0; r < 8; ++r)
          qk[(mt * 16 + hi16 + r) * 256 + cofs] = (__bf16)(acc[r] + bias);
      } else {                             // v: transposed, packed b128 store
        v8bf pk;
#pragma unroll
        for (int r = 0; r < 8; ++r) pk[r] = (__bf16)(acc[r] + bias);
        *(v8bf*)(vt + ((nt - 16) * 16 + lrow) * 64 + mt * 16 + hi16) = pk;
      }
    }
  }
  __syncthreads();

  // ---------- Stage 3: scores = q @ k^T, + bm table, softmax -------------
  const int hh = wvu >> 1;                 // head 0..3 (scalar)
  const int m0 = (wvu & 1) * 2;            // mtiles m0, m0+1
  const int variant = ((wi == 15) ? 2 : 0) | ((wj == 15) ? 1 : 0);
  const float* bmh = bm + (size_t)(variant * 4 + hh) * 4096;

  v8f S[2][4];
#pragma unroll
  for (int mi = 0; mi < 2; ++mi) {
    v16bf a = load_a_frag(qk, (m0 + mi) * 16 + lrow, 256, hh * 32, lane);
#pragma unroll
    for (int nt2 = 0; nt2 < 4; ++nt2) {
      v16bf bb = load_bt_frag(qk, nt2 * 16 + lrow, 256, 128 + hh * 32, lane);
      v8f acc = {};
      WMMA_BF16(acc, a, bb);
      S[mi][nt2] = acc;
    }
  }

  const int jcol = lrow;
#pragma unroll
  for (int mi = 0; mi < 2; ++mi) {
#pragma unroll
    for (int r = 0; r < 8; ++r) {
      const int i = (m0 + mi) * 16 + r + hi16;
      const v4f bv = *(const v4f*)(bmh + (i * 16 + jcol) * 4);
      float mx = -3.0e38f;
#pragma unroll
      for (int nt2 = 0; nt2 < 4; ++nt2) {
        const float s = S[mi][nt2][r] + bv[nt2];
        S[mi][nt2][r] = s;
        mx = fmaxf(mx, s);
      }
      for (int off = 1; off < 16; off <<= 1)
        mx = fmaxf(mx, __shfl_xor(mx, off, 32));
      float sum = 0.f;
#pragma unroll
      for (int nt2 = 0; nt2 < 4; ++nt2) {
        const float e = __expf(S[mi][nt2][r] - mx);
        S[mi][nt2][r] = e;
        sum += e;
      }
      for (int off = 1; off < 16; off <<= 1)
        sum += __shfl_xor(sum, off, 32);
      const float inv = 1.f / sum;
#pragma unroll
      for (int nt2 = 0; nt2 < 4; ++nt2) S[mi][nt2][r] *= inv;
    }
  }
  __syncthreads();                         // q/k reads done -> reuse as P

  __bf16* Pb = qk;                         // P: [4 heads][64][64] bf16
#pragma unroll
  for (int mi = 0; mi < 2; ++mi)
#pragma unroll
    for (int nt2 = 0; nt2 < 4; ++nt2)
#pragma unroll
      for (int r = 0; r < 8; ++r) {
        const int i = (m0 + mi) * 16 + r + hi16;
        Pb[hh * 4096 + i * 64 + nt2 * 16 + jcol] = (__bf16)S[mi][nt2][r];
      }
  __syncthreads();

  // ---------- Stage 4: attn-out = P @ V  (per head [64,64]x[64,32]) ------
  const int hh2 = wvu >> 1;
  const int nt3 = wvu & 1;
  v8f O[4];
#pragma unroll
  for (int mt = 0; mt < 4; ++mt) {
    v8f acc = {};
#pragma unroll
    for (int kk = 0; kk < 2; ++kk) {
      v16bf a = load_a_frag(Pb + hh2 * 4096, mt * 16 + lrow, 64, kk * 32, lane);
      v16bf bb = load_bt_frag(vt + hh2 * 2048, nt3 * 16 + lrow, 64, kk * 32, lane);
      WMMA_BF16(acc, a, bb);
    }
    O[mt] = acc;
  }
  __bf16* oa = xa;                         // reuse x tile as attn-out
#pragma unroll
  for (int mt = 0; mt < 4; ++mt)
#pragma unroll
    for (int r = 0; r < 8; ++r) {
      const int m = mt * 16 + r + hi16;
      oa[m * 128 + hh2 * 32 + nt3 * 16 + lrow] = (__bf16)O[mt][r];
    }
  __syncthreads();                         // P reads done -> reuse as rowoff

  // Row -> global-offset table for the rolled-back store (P region is free).
  int* rowoff = (int*)Pb;
  if (tid < 64) {
    int off = 0;
    if (tid < NTOK) {
      const int gh = (wi * 7 + (tid / 7) + SH) % H;
      const int gw = (wj * 7 + (tid % 7) + SH) % W;
      off = ((b * H + gh) * W + gw) * C;
    }
    rowoff[tid] = off;
  }
  __syncthreads();

  // ---------- Stage 5: out = attn-out @ proj^T + b, store (roll back) ----
  const int nt4 = wvu;
  v16bf Bp[4];
#pragma unroll
  for (int kk = 0; kk < 4; ++kk)
    Bp[kk] = load_bt_frag(projw_bf, nt4 * 16 + lrow, 128, kk * 32, lane);
  const int col = nt4 * 16 + lrow;
  const float pb = proj_b[col];
#pragma unroll
  for (int mt = 0; mt < 4; ++mt) {
    v8f acc = {};
#pragma unroll
    for (int kk = 0; kk < 4; ++kk) {
      v16bf a = load_a_frag(oa, mt * 16 + lrow, 128, kk * 32, lane);
      WMMA_BF16(acc, a, Bp[kk]);
    }
    const v4i r0 = *(const v4i*)(rowoff + mt * 16 + hi16);
    const v4i r1 = *(const v4i*)(rowoff + mt * 16 + hi16 + 4);
#pragma unroll
    for (int r = 0; r < 8; ++r) {
      const int roff = (r < 4) ? r0[r] : r1[r - 4];
      const int mlo = mt * 16 + r;         // compile-time per (mt,r)
      if (mlo + 8 < NTOK) {                // both lane-halves valid
        out[roff + col] = acc[r] + pb;
      } else if (mlo < NTOK) {             // only lanes 0-15 valid (mt=3,r=0)
        if (lane < 16) out[roff + col] = acc[r] + pb;
      }
    }
  }
}

extern "C" void kernel_launch(void* const* d_in, const int* in_sizes, int n_in,
                              void* d_out, int out_size, void* d_ws,
                              size_t ws_size, hipStream_t stream) {
  (void)in_sizes; (void)n_in; (void)ws_size; (void)out_size;
  const float* x      = (const float*)d_in[0];
  const float* qkv_w  = (const float*)d_in[1];
  const float* qkv_b  = (const float*)d_in[2];
  const float* proj_w = (const float*)d_in[3];
  const float* proj_b = (const float*)d_in[4];
  const float* reltab = (const float*)d_in[5];
  float* out = (float*)d_out;
  char* ws = (char*)d_ws;  // needs ~386 KB

  hipLaunchKernelGGL(swin_prep, dim3(512), dim3(256), 0, stream,
                     qkv_w, proj_w, qkv_b, reltab, ws);

  hipLaunchKernelGGL(swin_window_attn_fused, dim3(8192), dim3(256), 0, stream,
                     x, proj_b, (const char*)ws, out);
}
